// lstm_Bert_66915590472192
// MI455X (gfx1250) — matrix operand order repaired and checked by reading, not compiled, not run on hardware
//
#include <hip/hip_runtime.h>
#include <hip/hip_bf16.h>
#include <math.h>

// ---------------- problem constants ----------------
#define VOCAB   30522
#define VOCABP  30528    // padded to 16-row tiles -> unguarded WMMA epilogue stores
#define EMBED   768
#define H1      50
#define H2      25
#define LABELS  4
#define TLEN    64
#define NB      2048

// padded gate widths
#define D1STR   224      // per-direction layer-1 gate stride (4*H1=200 -> 14 tiles)
#define G1PAD   448      // both directions
#define D2STR   112      // per-direction layer-2 gate stride (4*H2=100 -> 7 tiles)
#define X2W     64       // packed bf16-pair words per row of layer-2 input (128 halves)

// ---------------- async global->LDS (gfx1250) with safe fallback ----------------
#if defined(__has_builtin)
#  if __has_builtin(__builtin_amdgcn_global_load_async_to_lds_b128)
#    define HAVE_ASYNC 1
#  endif
#endif
#ifndef HAVE_ASYNC
#  define HAVE_ASYNC 0
#endif

typedef int v4i_t __attribute__((vector_size(16)));
typedef __attribute__((address_space(1))) v4i_t* as1_v4i;
typedef __attribute__((address_space(3))) v4i_t* as3_v4i;

__device__ __forceinline__ void g2lds_b128(const void* g, void* l) {
#if HAVE_ASYNC
  __builtin_amdgcn_global_load_async_to_lds_b128(
      (as1_v4i)(unsigned long long)(size_t)g,
      (as3_v4i)(unsigned int)(size_t)l, 0, 0);
#else
  *(float4*)l = *(const float4*)g;
#endif
}

// wait until at most n async ops remain (completions are in-order)
#if HAVE_ASYNC
#  if defined(__has_builtin) && __has_builtin(__builtin_amdgcn_s_wait_asynccnt)
#    define WAIT_ASYNC(n) __builtin_amdgcn_s_wait_asynccnt(n)
#  else
#    define WAIT_ASYNC(n) asm volatile("s_wait_asynccnt %0" :: "i"(n) : "memory")
#  endif
#else
#  define WAIT_ASYNC(n) ((void)0)
#endif

// ---------------- fast transcendentals (CDNA5 V_TANH_F32) ----------------
#if defined(__has_builtin)
#  if __has_builtin(__builtin_amdgcn_tanhf)
#    define FAST_TANH(x) __builtin_amdgcn_tanhf(x)
#  elif __has_builtin(__builtin_amdgcn_tanh_f32)
#    define FAST_TANH(x) __builtin_amdgcn_tanh_f32(x)
#  endif
#endif
#ifndef FAST_TANH
#  define FAST_TANH(x) tanhf(x)
#endif

__device__ __forceinline__ float sigm(float x) {
  return 0.5f * FAST_TANH(0.5f * x) + 0.5f;     // sigma(x) = (1+tanh(x/2))/2
}

// ---------------- WMMA fragment types ----------------
typedef __attribute__((ext_vector_type(16))) __bf16 v16bf;
typedef __attribute__((ext_vector_type(8)))  float  v8f;

union FragBF { unsigned int u[8]; v16bf v; };

__device__ __forceinline__ v8f wmma_bf16(const FragBF& a, const FragBF& b, v8f c) {
  return __builtin_amdgcn_wmma_f32_16x16x32_bf16(false, a.v, false, b.v,
                                                 (short)0, c, false, false);
}

// A-matrix 16x32 bf16 layout: lane row = lane&15; VGPR v holds K-pair index:
__device__ __forceinline__ int a_kp(int v, int half) {
  return (v & 3) + ((v >> 2) << 3) + half * 4;
}
// B-matrix 32x16 bf16 layout: lane col = lane&15; VGPR v holds K-pair:
__device__ __forceinline__ int b_kp(int v, int half) {
  return (v & 3) + ((v >> 2) << 3) + half * 4;
}

// pack two f32 -> dword of 2 bf16 (RNE); single V_CVT_PK_BF16_F32 when available
#if defined(__has_builtin) && __has_builtin(__builtin_amdgcn_cvt_pk_bf16_f32)
__device__ __forceinline__ unsigned int bf16pair(float a, float b) {
  auto r = __builtin_amdgcn_cvt_pk_bf16_f32(a, b);
  unsigned int u;
  __builtin_memcpy(&u, &r, 4);
  return u;
}
#else
__device__ __forceinline__ unsigned int bf16pair(float a, float b) {
  unsigned int ua = __float_as_uint(a);
  unsigned int ub = __float_as_uint(b);
  ua = (ua + 0x7FFFu + ((ua >> 16) & 1u)) >> 16;
  ub = (ub + 0x7FFFu + ((ub >> 16) & 1u)) >> 16;
  return (ua & 0xFFFFu) | (ub << 16);
}
#endif

// monotonic float->uint map for atomicMax-based float max
__device__ __forceinline__ unsigned int encMax(float x) {
  unsigned int u = __float_as_uint(x);
  return u ^ ((unsigned int)((int)u >> 31) | 0x80000000u);
}
__device__ __forceinline__ float decMax(unsigned int k) {
  unsigned int u = (k & 0x80000000u) ? (k ^ 0x80000000u) : ~k;
  return __uint_as_float(u);
}

// ================= K0: weight repack (bf16 pair layouts) + pooled init ============
__device__ __forceinline__ float w2el(const float* W, int n, int k) {
  if (n >= 100) return 0.f;
  if (k < 64) return (k < 50) ? W[n * 100 + k] : 0.f;
  int kk = k - 64;
  return (kk < 50) ? W[n * 100 + 50 + kk] : 0.f;
}

__global__ __launch_bounds__(256)
void k0_prep(const float* __restrict__ w_ih1f, const float* __restrict__ w_ih1r,
             const float* __restrict__ w_hh1f, const float* __restrict__ w_hh1r,
             const float* __restrict__ w_ih2f, const float* __restrict__ w_ih2r,
             const float* __restrict__ w_hh2f, const float* __restrict__ w_hh2r,
             unsigned int* __restrict__ Wt1,   // [384][448]
             unsigned int* __restrict__ Whh1,  // [2][32][224]
             unsigned int* __restrict__ W2t,   // [2][64][112]
             unsigned int* __restrict__ Whh2,  // [2][16][112]
             unsigned int* __restrict__ pooled /* [64][64] */) {
  int idx = blockIdx.x * blockDim.x + threadIdx.x;
  if (idx < 384 * G1PAD) {
    int kp = idx / G1PAD, nn = idx % G1PAD;
    int d = nn / D1STR, n = nn % D1STR;
    const float* W = d ? w_ih1r : w_ih1f;
    unsigned int v = 0;
    if (n < 4 * H1) v = bf16pair(W[n * EMBED + 2 * kp], W[n * EMBED + 2 * kp + 1]);
    Wt1[idx] = v;
    return;
  }
  idx -= 384 * G1PAD;
  if (idx < 2 * 32 * D1STR) {
    int d = idx / (32 * D1STR); int r = idx % (32 * D1STR);
    int kp = r / D1STR, n = r % D1STR;
    const float* W = d ? w_hh1r : w_hh1f;
    int k = 2 * kp;
    float a = (n < 4 * H1 && k     < H1) ? W[n * H1 + k]     : 0.f;
    float b = (n < 4 * H1 && k + 1 < H1) ? W[n * H1 + k + 1] : 0.f;
    Whh1[idx] = bf16pair(a, b);
    return;
  }
  idx -= 2 * 32 * D1STR;
  if (idx < 2 * 64 * D2STR) {
    int d = idx / (64 * D2STR); int r = idx % (64 * D2STR);
    int kp = r / D2STR, n = r % D2STR;
    const float* W = d ? w_ih2r : w_ih2f;
    W2t[idx] = bf16pair(w2el(W, n, 2 * kp), w2el(W, n, 2 * kp + 1));
    return;
  }
  idx -= 2 * 64 * D2STR;
  if (idx < 2 * 16 * D2STR) {
    int d = idx / (16 * D2STR); int r = idx % (16 * D2STR);
    int kp = r / D2STR, n = r % D2STR;
    const float* W = d ? w_hh2r : w_hh2f;
    int k = 2 * kp;
    float a = (n < 4 * H2 && k     < H2) ? W[n * H2 + k]     : 0.f;
    float b = (n < 4 * H2 && k + 1 < H2) ? W[n * H2 + k + 1] : 0.f;
    Whh2[idx] = bf16pair(a, b);
    return;
  }
  idx -= 2 * 16 * D2STR;
  if (idx < TLEN * 64) pooled[idx] = 0u;
}

// ================= K1: vocab projection  P = E @ W1^T  (WMMA bf16) ============
// B panel double-buffered in LDS; issue panel s+1, then wait <=14 for panel s.
__global__ __launch_bounds__(128)
void k1_vocab_proj(const float* __restrict__ embed,
                   const unsigned int* __restrict__ Wt1,   // [384][448]
                   float* __restrict__ P /* [VOCABP][448] */) {
  __shared__ __align__(16) unsigned int Bs[2][16][G1PAD];

  const int tid = threadIdx.x;
  const int mtile = blockIdx.x;
  const int wave = tid >> 5, lane = tid & 31;
  const int half = lane >> 4, lrow = lane & 15;
  const int row = mtile * 16 + lrow;
  const int rowc = row < VOCAB ? row : VOCAB - 1;
  const float2* e2 = reinterpret_cast<const float2*>(embed) + (size_t)rowc * (EMBED / 2);

  auto load_panel = [&](int s) {               // 14 async ops per thread
    const int buf = s & 1;
    for (int i = tid; i < 16 * 112; i += 128) {
      int kpl = i / 112, q = (i % 112) * 4;
      g2lds_b128(Wt1 + (size_t)(s * 16 + kpl) * G1PAD + q, &Bs[buf][kpl][q]);
    }
  };

  load_panel(0);

  v8f acc[7];
#pragma unroll
  for (int i = 0; i < 7; ++i) acc[i] = (v8f){0.f,0.f,0.f,0.f,0.f,0.f,0.f,0.f};

  for (int s = 0; s < EMBED / 32; ++s) {
    if (s + 1 < EMBED / 32) { load_panel(s + 1); WAIT_ASYNC(14); }
    else                    { WAIT_ASYNC(0); }
    __syncthreads();                          // panel s visible to all waves
    __builtin_prefetch((const void*)(e2 + (s + 1) * 16), 0, 0);
    FragBF A;
#pragma unroll
    for (int v = 0; v < 8; ++v) {
      float2 ab = e2[s * 16 + a_kp(v, half)];
      A.u[v] = bf16pair(ab.x, ab.y);
    }
    const int buf = s & 1;
#pragma unroll
    for (int i = 0; i < 7; ++i) {
      const int tile = wave * 7 + i;          // 0..27
      FragBF B;
#pragma unroll
      for (int v = 0; v < 8; ++v)
        B.u[v] = Bs[buf][b_kp(v, half)][tile * 16 + lrow];
      acc[i] = wmma_bf16(A, B, acc[i]);
    }
    __syncthreads();                          // reads of panel s done
  }
#pragma unroll
  for (int i = 0; i < 7; ++i) {
    const int col = (wave * 7 + i) * 16 + lrow;
#pragma unroll
    for (int r = 0; r < 8; ++r)               // P padded: no guard needed
      P[(size_t)(mtile * 16 + r + half * 8) * G1PAD + col] = acc[i][r];
  }
}

// ================= K2: layer-1 BiLSTM recurrence ============
// Whh slab resident in LDS; token-gather double-buffered: step t+1's xg is in
// flight (async) while step t computes.  h kept packed (hp) for raw A loads.
__global__ __launch_bounds__(128)
void k2_lstm1(const int* __restrict__ inputs,
              const float* __restrict__ P,            // [VOCABP][448]
              const unsigned int* __restrict__ Whh1,  // [2][32][224]
              const float* __restrict__ b1f, const float* __restrict__ b1r,
              unsigned int* __restrict__ X2pair /* [T*N][64] */) {
  __shared__ __align__(16) unsigned int Ws[32][D1STR];
  __shared__ __align__(16) float G[2][16][D1STR];
  __shared__ float hbuf[16][64];              // f32 h (zero-padded to 64)
  __shared__ unsigned int hp[16][32];         // packed bf16 pairs of hbuf
  __shared__ float cbuf[16][H1];

  const int dir = blockIdx.x & 1;
  const int n0 = (blockIdx.x >> 1) * 16;
  const float* bias = dir ? b1r : b1f;
  const int tid = threadIdx.x;
  const int wave = tid >> 5, lane = tid & 31, half = lane >> 4, lrow = lane & 15;

  auto prefill = [&](int teStep, int buf) {   // 7 async ops per thread
    for (int i = tid; i < 16 * 56; i += 128) {
      int m = i / 56, q = (i % 56) * 4;
      int token = inputs[teStep * NB + n0 + m];
      g2lds_b128(P + (size_t)token * G1PAD + dir * D1STR + q, &G[buf][m][q]);
    }
  };

  // stage Whh direction slab once (32 x 224 dwords) + first gate tile
  for (int i = tid; i < 32 * 56; i += 128) {
    int kp = i / 56, q = (i % 56) * 4;
    g2lds_b128(Whh1 + (size_t)(dir * 32 + kp) * D1STR + q, &Ws[kp][q]);
  }
  prefill(dir ? TLEN - 1 : 0, 0);
  for (int i = tid; i < 16 * 64; i += 128) hbuf[i >> 6][i & 63] = 0.f;
  for (int i = tid; i < 16 * 32; i += 128) hp[i >> 5][i & 31] = 0u;
  for (int i = tid; i < 16 * H1; i += 128) cbuf[i / H1][i % H1] = 0.f;
  WAIT_ASYNC(0);
  __syncthreads();

  for (int t = 0; t < TLEN; ++t) {
    const int te = dir ? (TLEN - 1 - t) : t;
    const int cur = t & 1, nxt = cur ^ 1;

    // launch next step's gather into the other buffer, then wait for this one
    if (t + 1 < TLEN) { prefill(dir ? te - 1 : te + 1, nxt); WAIT_ASYNC(7); }
    else              { WAIT_ASYNC(0); }

    FragBF A0, A1;                               // h fragment, K 0..31 / 32..63
#pragma unroll
    for (int v = 0; v < 8; ++v) {
      int kp = a_kp(v, half);
      A0.u[v] = hp[lrow][kp];
      A1.u[v] = hp[lrow][16 + kp];
    }
    __syncthreads();                             // G[cur] + hp stable

    for (int tile = wave; tile < 13; tile += 4) {
      const int col = tile * 16 + lrow;
      v8f acc = (v8f){0.f,0.f,0.f,0.f,0.f,0.f,0.f,0.f};
      FragBF B0, B1;
#pragma unroll
      for (int v = 0; v < 8; ++v) {
        int kpl = b_kp(v, half);
        B0.u[v] = Ws[kpl][col];
        B1.u[v] = Ws[16 + kpl][col];
      }
      acc = wmma_bf16(A0, B0, acc);
      acc = wmma_bf16(A1, B1, acc);
#pragma unroll
      for (int r = 0; r < 8; ++r) G[cur][r + half * 8][col] += acc[r];
    }
    __syncthreads();

    for (int u = tid; u < 16 * H1; u += 128) {   // gate nonlinearity + state update
      int m = u / H1, j = u % H1;
      float gi = G[cur][m][j]          + bias[j];
      float gf = G[cur][m][H1 + j]     + bias[H1 + j];
      float gg = G[cur][m][2 * H1 + j] + bias[2 * H1 + j];
      float go = G[cur][m][3 * H1 + j] + bias[3 * H1 + j];
      float c = sigm(gf) * cbuf[m][j] + sigm(gi) * FAST_TANH(gg);
      cbuf[m][j] = c;
      hbuf[m][j] = sigm(go) * FAST_TANH(c);
    }
    __syncthreads();

    for (int u = tid; u < 16 * 32; u += 128) {   // pack h once; doubles as X2 emit
      int m = u >> 5, w = u & 31;
      unsigned int pv = bf16pair(hbuf[m][2 * w], hbuf[m][2 * w + 1]);
      hp[m][w] = pv;
      X2pair[(size_t)(te * NB + n0 + m) * X2W + dir * 32 + w] = pv;
    }
    __syncthreads();
  }
}

// ================= K3: layer-2 BiLSTM (fused input GEMM) + max-pool ============
__global__ __launch_bounds__(128)
void k3_lstm2(const unsigned int* __restrict__ X2pair,  // [T*N][64]
              const unsigned int* __restrict__ W2t,     // [2][64][112]
              const unsigned int* __restrict__ Whh2,    // [2][16][112]
              const float* __restrict__ b2f, const float* __restrict__ b2r,
              unsigned int* __restrict__ pooled /* [64][64] */) {
  __shared__ __align__(16) unsigned int W2s[64][D2STR];
  __shared__ __align__(16) unsigned int Whh2s[16][D2STR];
  __shared__ __align__(16) unsigned int X2s[2][16][X2W];
  __shared__ float G[16][D2STR];
  __shared__ float hbuf[16][32];              // f32 h (zero-padded to 32)
  __shared__ unsigned int hp2[16][16];        // packed bf16 pairs of hbuf
  __shared__ float cbuf[16][H2];

  const int dir = blockIdx.x & 1;
  const int n0 = (blockIdx.x >> 1) * 16;
  const float* bias = dir ? b2r : b2f;
  const int tid = threadIdx.x;
  const int wave = tid >> 5, lane = tid & 31, half = lane >> 4, lrow = lane & 15;

  auto stageX = [&](int teStep, int buf) {    // 2 async ops per thread
    for (int i = tid; i < 16 * 16; i += 128) {
      int m = i >> 4, q = (i & 15) * 4;
      g2lds_b128(X2pair + (size_t)(teStep * NB + n0 + m) * X2W + q, &X2s[buf][m][q]);
    }
  };

  // stage weight slabs once + first activation tile
  for (int i = tid; i < 64 * 28; i += 128) {
    int kp = i / 28, q = (i % 28) * 4;
    g2lds_b128(W2t + (size_t)(dir * 64 + kp) * D2STR + q, &W2s[kp][q]);
  }
  for (int i = tid; i < 16 * 28; i += 128) {
    int kp = i / 28, q = (i % 28) * 4;
    g2lds_b128(Whh2 + (size_t)(dir * 16 + kp) * D2STR + q, &Whh2s[kp][q]);
  }
  stageX(dir ? TLEN - 1 : 0, 0);
  for (int i = tid; i < 16 * 32; i += 128) hbuf[i >> 5][i & 31] = 0.f;
  for (int i = tid; i < 16 * 16; i += 128) hp2[i >> 4][i & 15] = 0u;
  for (int i = tid; i < 16 * H2; i += 128) cbuf[i / H2][i % H2] = 0.f;
  WAIT_ASYNC(0);
  __syncthreads();

  for (int t = 0; t < TLEN; ++t) {
    const int te = dir ? (TLEN - 1 - t) : t;
    const int cur = t & 1, nxt = cur ^ 1;

    if (t + 1 < TLEN) { stageX(dir ? te - 1 : te + 1, nxt); WAIT_ASYNC(2); }
    else              { WAIT_ASYNC(0); }

    FragBF AH;                                    // recurrent fragment (K=32)
#pragma unroll
    for (int v = 0; v < 8; ++v) AH.u[v] = hp2[lrow][a_kp(v, half)];
    __syncthreads();                              // X2s[cur] + hp2 stable

    FragBF AX[4];                                 // layer-2 input fragments (K=128)
#pragma unroll
    for (int s = 0; s < 4; ++s)
#pragma unroll
      for (int v = 0; v < 8; ++v)
        AX[s].u[v] = X2s[cur][lrow][s * 16 + a_kp(v, half)];

    for (int tile = wave; tile < 7; tile += 4) {
      const int col = tile * 16 + lrow;
      v8f acc = (v8f){0.f,0.f,0.f,0.f,0.f,0.f,0.f,0.f};
#pragma unroll
      for (int s = 0; s < 4; ++s) {
        FragBF B;
#pragma unroll
        for (int v = 0; v < 8; ++v) B.u[v] = W2s[s * 16 + b_kp(v, half)][col];
        acc = wmma_bf16(AX[s], B, acc);
      }
      {
        FragBF B;
#pragma unroll
        for (int v = 0; v < 8; ++v) B.u[v] = Whh2s[b_kp(v, half)][col];
        acc = wmma_bf16(AH, B, acc);
      }
#pragma unroll
      for (int r = 0; r < 8; ++r) G[r + half * 8][col] = acc[r];
    }
    __syncthreads();

    for (int u = tid; u < 16 * H2; u += 128) {
      int m = u / H2, j = u % H2;
      float gi = G[m][j]          + bias[j];
      float gf = G[m][H2 + j]     + bias[H2 + j];
      float gg = G[m][2 * H2 + j] + bias[2 * H2 + j];
      float go = G[m][3 * H2 + j] + bias[3 * H2 + j];
      float c = sigm(gf) * cbuf[m][j] + sigm(gi) * FAST_TANH(gg);
      cbuf[m][j] = c;
      hbuf[m][j] = sigm(go) * FAST_TANH(c);
    }
    __syncthreads();

    for (int u = tid; u < 16 * 16; u += 128)       // pack h once per step
      hp2[u >> 4][u & 15] = bf16pair(hbuf[u >> 4][2 * (u & 15)],
                                     hbuf[u >> 4][2 * (u & 15) + 1]);
    if (tid < H2) {                                // partial batch max -> global max
      float mx = hbuf[0][tid];
#pragma unroll
      for (int m = 1; m < 16; ++m) mx = fmaxf(mx, hbuf[m][tid]);
      atomicMax(&pooled[te * 64 + dir * H2 + tid], encMax(mx));
    }
    __syncthreads();
  }
}

// ================= K4: FC head + softmax ============
__global__ __launch_bounds__(64)
void k4_head(const unsigned int* __restrict__ pooled,
             const float* __restrict__ fc1_w, const float* __restrict__ fc1_b,
             const float* __restrict__ fc2_w, const float* __restrict__ fc2_b,
             float* __restrict__ out /* [64][4] */) {
  const int t = threadIdx.x;
  float x[2 * H2];
#pragma unroll
  for (int f = 0; f < 2 * H2; ++f) x[f] = decMax(pooled[t * 64 + f]);
  float y[2 * H2];
  for (int g = 0; g < 2 * H2; ++g) {
    float s = fc1_b[g];
    for (int f = 0; f < 2 * H2; ++f) s += x[f] * fc1_w[g * (2 * H2) + f];
    y[g] = fmaxf(s, 0.f);
  }
  float z[LABELS];
  float zmax = -1e30f;
#pragma unroll
  for (int l = 0; l < LABELS; ++l) {
    float s = fc2_b[l];
    for (int g = 0; g < 2 * H2; ++g) s += y[g] * fc2_w[l * (2 * H2) + g];
    z[l] = s;
    zmax = fmaxf(zmax, s);
  }
  float sum = 0.f;
#pragma unroll
  for (int l = 0; l < LABELS; ++l) { z[l] = __expf(z[l] - zmax); sum += z[l]; }
#pragma unroll
  for (int l = 0; l < LABELS; ++l) out[t * LABELS + l] = z[l] / sum;
}

// ================= launch ============
extern "C" void kernel_launch(void* const* d_in, const int* in_sizes, int n_in,
                              void* d_out, int out_size, void* d_ws, size_t ws_size,
                              hipStream_t stream) {
  const int*   inputs = (const int*)  d_in[0];
  const float* embed  = (const float*)d_in[1];
  const float* w_ih1f = (const float*)d_in[2];
  const float* w_hh1f = (const float*)d_in[3];
  const float* b1f    = (const float*)d_in[4];
  const float* w_ih1r = (const float*)d_in[5];
  const float* w_hh1r = (const float*)d_in[6];
  const float* b1r    = (const float*)d_in[7];
  const float* w_ih2f = (const float*)d_in[8];
  const float* w_hh2f = (const float*)d_in[9];
  const float* b2f    = (const float*)d_in[10];
  const float* w_ih2r = (const float*)d_in[11];
  const float* w_hh2r = (const float*)d_in[12];
  const float* b2r    = (const float*)d_in[13];
  const float* fc1_w  = (const float*)d_in[14];
  const float* fc1_b  = (const float*)d_in[15];
  const float* fc2_w  = (const float*)d_in[16];
  const float* fc2_b  = (const float*)d_in[17];

  unsigned int* ws = (unsigned int*)d_ws;
  size_t o = 0;
  float*        P      = (float*)(ws + o); o += (size_t)VOCABP * G1PAD;       // 54.7 MB
  unsigned int* X2pair = ws + o;           o += (size_t)TLEN * NB * X2W;      // 33.6 MB
  unsigned int* Wt1    = ws + o;           o += 384 * G1PAD;
  unsigned int* Whh1   = ws + o;           o += 2 * 32 * D1STR;
  unsigned int* W2t    = ws + o;           o += 2 * 64 * D2STR;
  unsigned int* Whh2   = ws + o;           o += 2 * 16 * D2STR;
  unsigned int* pooled = ws + o;           o += TLEN * 64;

  const int prepN = 384 * G1PAD + 2 * 32 * D1STR + 2 * 64 * D2STR + 2 * 16 * D2STR + TLEN * 64;
  k0_prep<<<(prepN + 255) / 256, 256, 0, stream>>>(
      w_ih1f, w_ih1r, w_hh1f, w_hh1r, w_ih2f, w_ih2r, w_hh2f, w_hh2r,
      Wt1, Whh1, W2t, Whh2, pooled);

  k1_vocab_proj<<<VOCABP / 16, 128, 0, stream>>>(embed, Wt1, P);

  k2_lstm1<<<(NB / 16) * 2, 128, 0, stream>>>(inputs, P, Whh1, b1f, b1r, X2pair);

  k3_lstm2<<<(NB / 16) * 2, 128, 0, stream>>>(X2pair, W2t, Whh2, b2f, b2r, pooled);

  k4_head<<<1, 64, 0, stream>>>(pooled, fc1_w, fc1_b, fc2_w, fc2_b, (float*)d_out);
}
